// BidirectionalMambaBlock_13958643712580
// MI455X (gfx1250) — matrix operand order repaired and checked by reading, not compiled, not run on hardware
//
#include <hip/hip_runtime.h>
#include <hip/hip_bf16.h>

// ---------------------------------------------------------------------------
// Bidirectional Mamba block for gfx1250 (MI455X).
// GEMMs: WMMA bf16 16x16x32 (fp32 accum), bf16 activations end-to-end.
// Tile staging via Tensor Data Mover (TENSOR_LOAD_TO_LDS + s_wait_tensorcnt)
// when available; B128 manual staging fallback otherwise.
// ---------------------------------------------------------------------------

typedef __attribute__((ext_vector_type(16))) __bf16 v16bf;
typedef __attribute__((ext_vector_type(8)))  float  v8f;
typedef unsigned int v4u __attribute__((ext_vector_type(4)));
typedef int          v8i __attribute__((ext_vector_type(8)));
typedef int          v4i __attribute__((ext_vector_type(4)));

#if __has_builtin(__builtin_amdgcn_tensor_load_to_lds) && \
    __has_builtin(__builtin_amdgcn_s_wait_tensorcnt)
#define HAVE_TDM 1
#else
#define HAVE_TDM 0
#endif

#define D_MODEL 512
#define D_INNER 1024
#define DT_RANK 32
#define D_STATE 16
#define BATCH   2
#define SEQ     1024
#define ROWS    (BATCH * SEQ)   // 2048

#define BM 64
#define BN 64
#define BK 32
#define LDS_STRIDE 40           // padded LDS row stride (bf16), 80B = 5x16B

// -------------------------- fp32 -> bf16 cast (x4) -------------------------
__global__ void k_cast4(const float* __restrict__ src, __bf16* __restrict__ dst, int n4) {
    int i = blockIdx.x * 256 + threadIdx.x;
    if (i < n4) {
        float4 f = ((const float4*)src)[i];
        __bf16 o[4] = { (__bf16)f.x, (__bf16)f.y, (__bf16)f.z, (__bf16)f.w };
        *(uint2*)(dst + 4 * (size_t)i) = *(const uint2*)o;
    }
}

#if HAVE_TDM
// Issue one TDM 2D-tile load: tile_rows x tile_k bf16 elems from a row-major
// matrix (row stride = stride_elems) into LDS at lds_addr, padding each 64B
// row with 16B so the LDS layout has stride LDS_STRIDE(=40) bf16 elems.
__device__ __forceinline__ void tdm_load_tile(unsigned lds_addr,
                                              const __bf16* gtile,
                                              unsigned row_len_elems,
                                              unsigned nrows,
                                              unsigned stride_elems,
                                              unsigned tile_k,
                                              unsigned tile_rows)
{
    unsigned long long ga = (unsigned long long)(uintptr_t)gtile;
    v4u g0;
    g0.x = 1u;                                      // count=1, user descriptor
    g0.y = lds_addr;                                // LDS byte address
    g0.z = (unsigned)(ga & 0xFFFFFFFFu);            // global_addr[31:0]
    g0.w = (unsigned)((ga >> 32) & 0x01FFFFFFu)     // global_addr[56:32]
         | (2u << 30);                              // type = 2 ("image")
    unsigned d0 = (1u << 16)                        // data_size = 2 bytes
                | (1u << 20)                        // pad_enable
                | (3u << 22)                        // pad_interval = 16 DWORDs
                | (3u << 25);                       // pad_amount   = 4 DWORDs
    unsigned long long st = (unsigned long long)stride_elems;
    v8i g1;
    g1[0] = (int)d0;
    g1[1] = (int)((row_len_elems & 0xFFFFu) << 16);                 // dim0[15:0]
    g1[2] = (int)(((row_len_elems >> 16) & 0xFFFFu)
                | ((nrows & 0xFFFFu) << 16));                       // dim0 hi | dim1 lo
    g1[3] = (int)(((nrows >> 16) & 0xFFFFu)
                | ((tile_k & 0xFFFFu) << 16));                      // dim1 hi | tile_dim0
    g1[4] = (int)(tile_rows & 0xFFFFu);                             // tile_dim1 (tile_dim2=0)
    g1[5] = (int)(unsigned)(st & 0xFFFFFFFFu);                      // dim0_stride lo
    g1[6] = (int)(unsigned)((st >> 32) & 0xFFFFu);                  // dim0_stride hi
    g1[7] = 0;
    v4i z4 = { 0, 0, 0, 0 };
    v8i z8 = { 0, 0, 0, 0, 0, 0, 0, 0 };
    __builtin_amdgcn_tensor_load_to_lds(g0, g1, z4, z4, z8, 0);
}
#endif

// ------------------------------ WMMA GEMM ----------------------------------
// D[M x N](bf16) = act( A[M x K](bf16, lda) @ W[N x K]^T(bf16) + bias )
// act: 0 = none, 1 = relu, 2 = softplus
__global__ __launch_bounds__(256)
void k_gemm(const __bf16* __restrict__ A, int lda,
            const __bf16* __restrict__ W,
            const float* __restrict__ bias,
            __bf16* __restrict__ D, int ldd,
            int K, int act)
{
    __shared__ __bf16 As[BM * LDS_STRIDE];
    __shared__ __bf16 Bs[BN * LDS_STRIDE];

    const int tid  = threadIdx.x;
    const int lane = tid & 31;
    const int wave = tid >> 5;
    const int wm   = wave & 3;          // 4 waves along M
    const int wn   = wave >> 2;         // 2 waves along N (each does 2 WMMAs)
    const int m0   = blockIdx.y * BM;
    const int n0   = blockIdx.x * BN;
    const int hi   = lane >> 4;
    const int lm   = lane & 15;

    v8f acc0 = {}, acc1 = {};

#if HAVE_TDM
    const unsigned ldsA = (unsigned)(uintptr_t)(&As[0]);
    const unsigned ldsB = (unsigned)(uintptr_t)(&Bs[0]);

    for (int k0 = 0; k0 < K; k0 += BK) {
        if (wave == 0) {
            tdm_load_tile(ldsA, A + (size_t)m0 * lda + k0,
                          (unsigned)lda, (unsigned)ROWS, (unsigned)lda, BK, BM);
            tdm_load_tile(ldsB, W + (size_t)n0 * K + k0,
                          (unsigned)K, 4096u, (unsigned)K, BK, BN);
            __builtin_amdgcn_s_wait_tensorcnt(0);
        }
        __syncthreads();
#else
    // Manual staging role: one 16B chunk per thread per operand tile.
    const int sr = tid >> 2;            // 0..63  (tile row)
    const int sc = (tid & 3) * 8;       // 0,8,16,24 (tile col, bf16 elems)
    const __bf16* Aptr = A + (size_t)(m0 + sr) * lda + sc;
    const __bf16* Wptr = W + (size_t)(n0 + sr) * K + sc;

    for (int k0 = 0; k0 < K; k0 += BK) {
        uint4 av = *(const uint4*)(Aptr + k0);
        uint4 bv = *(const uint4*)(Wptr + k0);
        if (k0 + BK < K) {
            __builtin_prefetch(Aptr + k0 + BK, 0, 3);
            __builtin_prefetch(Wptr + k0 + BK, 0, 3);
        }
        *(uint4*)(&As[sr * LDS_STRIDE + sc]) = av;
        *(uint4*)(&Bs[sr * LDS_STRIDE + sc]) = bv;
        __syncthreads();
#endif
        // Fragment gather: contiguous k = 16*hi + i, same k order for A and B.
        v16bf a, b0, b1;
        #pragma unroll
        for (int i = 0; i < 16; ++i) {
            int kA = (hi << 4) + i;                           // A 16x32
            a[i]  = As[(wm * 16 + lm) * LDS_STRIDE + kA];
            int kB = (hi << 4) + i;                           // B 32x16
            b0[i] = Bs[(wn * 32 + lm) * LDS_STRIDE + kB];
            b1[i] = Bs[(wn * 32 + 16 + lm) * LDS_STRIDE + kB];
        }
        acc0 = __builtin_amdgcn_wmma_f32_16x16x32_bf16(
                   false, a, false, b0, (short)0, acc0, false, false);
        acc1 = __builtin_amdgcn_wmma_f32_16x16x32_bf16(
                   false, a, false, b1, (short)0, acc1, false, false);
        __syncthreads();
    }

    // Epilogue: C/D layout -> M = r + 8*hi, N = lm.
    #pragma unroll
    for (int r = 0; r < 8; ++r) {
        int row  = m0 + wm * 16 + r + hi * 8;
        int col0 = n0 + wn * 32 + lm;
        int col1 = col0 + 16;
        float v0 = acc0[r], v1 = acc1[r];
        if (bias) { v0 += bias[col0]; v1 += bias[col1]; }
        if (act == 1) { v0 = fmaxf(v0, 0.0f); v1 = fmaxf(v1, 0.0f); }
        else if (act == 2) {
            v0 = (v0 > 20.0f) ? v0 : logf(1.0f + __expf(v0));
            v1 = (v1 > 20.0f) ? v1 : logf(1.0f + __expf(v1));
        }
        D[(size_t)row * ldd + col0] = (__bf16)v0;
        D[(size_t)row * ldd + col1] = (__bf16)v1;
    }
}

// ----------------------- depthwise causal conv + SiLU ----------------------
__global__ void k_conv_silu(const __bf16* __restrict__ xz,   // [ROWS][2048]
                            const float* __restrict__ cw,    // [D_INNER][4]
                            const float* __restrict__ cb,    // [D_INNER]
                            __bf16* __restrict__ xm)         // [ROWS][1024]
{
    int i = blockIdx.x * 256 + threadIdx.x;      // over ROWS*D_INNER
    int d = i & (D_INNER - 1);
    int r = i >> 10;
    int t = r & (SEQ - 1);
    int b = r >> 10;
    float s = cb[d];
    #pragma unroll
    for (int j = 0; j < 4; ++j) {
        int tt = t - 3 + j;
        if (tt >= 0)
            s += (float)xz[(size_t)(b * SEQ + tt) * (2 * D_INNER) + d] * cw[d * 4 + j];
    }
    float sig = 1.0f / (1.0f + __expf(-s));
    xm[i] = (__bf16)(s * sig);
}

// --------------------------- time-flip [B,W,512] bf16 ----------------------
__global__ void k_flip512(const __bf16* __restrict__ src, __bf16* __restrict__ dst)
{
    int i = blockIdx.x * 256 + threadIdx.x;      // over ROWS*512/8 uint4 chunks
    int v = i & 63;                              // 64 x 8 elems per row
    int r = i >> 6;
    int t = r & (SEQ - 1);
    int b = r >> 10;
    uint4 d4 = ((const uint4*)src)[(size_t)r * 64 + v];
    ((uint4*)dst)[(size_t)(b * SEQ + (SEQ - 1 - t)) * 64 + v] = d4;
}

// ----------------------------- selective scan ------------------------------
// grid: (D_INNER/256, BATCH). One thread = one inner channel; h[16] in regs.
// Bm/Cm staged to LDS in 64-timestep chunks (one uint4 per thread per chunk).
#define SCAN_CH 64
__global__ __launch_bounds__(256)
void k_scan(const __bf16* __restrict__ delta,   // [ROWS][D_INNER]
            const __bf16* __restrict__ dbc,     // [ROWS][64] (dt|B|C)
            const __bf16* __restrict__ xm,      // [ROWS][D_INNER]
            const __bf16* __restrict__ xz,      // [ROWS][2048] (z at +1024)
            const float* __restrict__ A_log,    // [D_INNER][D_STATE]
            const float* __restrict__ Dp,       // [D_INNER]
            __bf16* __restrict__ y)             // [ROWS][D_INNER]
{
    int d = blockIdx.x * 256 + threadIdx.x;
    int b = blockIdx.y;

    float Av[D_STATE];
    #pragma unroll
    for (int s = 0; s < D_STATE; ++s) Av[s] = -__expf(A_log[d * D_STATE + s]);
    float Dd = Dp[d];
    float h[D_STATE];
    #pragma unroll
    for (int s = 0; s < D_STATE; ++s) h[s] = 0.0f;

    __shared__ __bf16 sBC[SCAN_CH][32];          // per step: B[0..15] | C[0..15]

    const int rr = threadIdx.x >> 2;             // 0..63
    const int cc = (threadIdx.x & 3) * 8;        // 0..24

    for (int t0 = 0; t0 < SEQ; t0 += SCAN_CH) {
        __syncthreads();
        {
            size_t r = (size_t)(b * SEQ + t0 + rr);
            *(uint4*)&sBC[rr][cc] = *(const uint4*)(dbc + r * 64 + DT_RANK + cc);
        }
        __syncthreads();

        for (int tt = 0; tt < SCAN_CH; ++tt) {
            size_t r = (size_t)(b * SEQ + t0 + tt);
            float dt = (float)delta[r * D_INNER + d];
            float x  = (float)xm[r * D_INNER + d];
            float zz = (float)xz[r * (2 * D_INNER) + D_INNER + d];

            float yv = 0.0f;
            #pragma unroll
            for (int s = 0; s < D_STATE; ++s) {
                float dA = __expf(dt * Av[s]);
                h[s] = dA * h[s] + dt * (float)sBC[tt][s] * x;
                yv += h[s] * (float)sBC[tt][16 + s];
            }
            yv += x * Dd;
            float sig = 1.0f / (1.0f + __expf(-zz));
            y[r * D_INNER + d] = (__bf16)(yv * (zz * sig));
        }
    }
}

// ------------------------- residual add + LayerNorm ------------------------
__global__ __launch_bounds__(256)
void k_add_ln(const __bf16* __restrict__ X, const __bf16* __restrict__ R,
              const float* __restrict__ g, const float* __restrict__ bb,
              __bf16* __restrict__ outB, float* __restrict__ outF, int accum)
{
    __shared__ float s1[256];
    __shared__ float s2[256];
    int row = blockIdx.x;
    int tid = threadIdx.x;
    const __bf16* xr = X + (size_t)row * D_MODEL;
    const __bf16* rr = R + (size_t)row * D_MODEL;

    float v0 = (float)xr[tid] + (float)rr[tid];
    float v1 = (float)xr[tid + 256] + (float)rr[tid + 256];
    s1[tid] = v0 + v1;
    s2[tid] = v0 * v0 + v1 * v1;
    __syncthreads();
    #pragma unroll
    for (int off = 128; off > 0; off >>= 1) {
        if (tid < off) { s1[tid] += s1[tid + off]; s2[tid] += s2[tid + off]; }
        __syncthreads();
    }
    float mean = s1[0] * (1.0f / D_MODEL);
    float var  = s2[0] * (1.0f / D_MODEL) - mean * mean;
    float rstd = rsqrtf(var + 1e-5f);

    float o0 = (v0 - mean) * rstd * g[tid] + bb[tid];
    float o1 = (v1 - mean) * rstd * g[tid + 256] + bb[tid + 256];

    if (outF) {
        float* op = outF + (size_t)row * D_MODEL;
        op[tid] = o0; op[tid + 256] = o1;
    } else {
        __bf16* op = outB + (size_t)row * D_MODEL;
        if (accum) {
            op[tid]       = (__bf16)((float)op[tid] + o0);
            op[tid + 256] = (__bf16)((float)op[tid + 256] + o1);
        } else {
            op[tid] = (__bf16)o0; op[tid + 256] = (__bf16)o1;
        }
    }
}

// ---------------------------------------------------------------------------
extern "C" void kernel_launch(void* const* d_in, const int* in_sizes, int n_in,
                              void* d_out, int out_size, void* d_ws, size_t ws_size,
                              hipStream_t stream)
{
    const float* x        = (const float*)d_in[0];
    const float* fm_in_W  = (const float*)d_in[1];
    const float* fm_cw    = (const float*)d_in[2];
    const float* fm_cb    = (const float*)d_in[3];
    const float* fm_xp_W  = (const float*)d_in[4];
    const float* fm_dt_W  = (const float*)d_in[5];
    const float* fm_dt_b  = (const float*)d_in[6];
    const float* fm_Alog  = (const float*)d_in[7];
    const float* fm_D     = (const float*)d_in[8];
    const float* fm_out_W = (const float*)d_in[9];
    const float* bm_in_W  = (const float*)d_in[10];
    const float* bm_cw    = (const float*)d_in[11];
    const float* bm_cb    = (const float*)d_in[12];
    const float* bm_xp_W  = (const float*)d_in[13];
    const float* bm_dt_W  = (const float*)d_in[14];
    const float* bm_dt_b  = (const float*)d_in[15];
    const float* bm_Alog  = (const float*)d_in[16];
    const float* bm_D     = (const float*)d_in[17];
    const float* bm_out_W = (const float*)d_in[18];
    const float* ln1_g = (const float*)d_in[19];
    const float* ln1_b = (const float*)d_in[20];
    const float* ln2_g = (const float*)d_in[21];
    const float* ln2_b = (const float*)d_in[22];
    const float* ln3_g = (const float*)d_in[23];
    const float* ln3_b = (const float*)d_in[24];
    const float* ff_W1 = (const float*)d_in[25];
    const float* ff_b1 = (const float*)d_in[26];
    const float* ff_W2 = (const float*)d_in[27];
    const float* ff_b2 = (const float*)d_in[28];

    // ---------------- workspace layout (all bf16) ----------------
    char* p = (char*)d_ws;
    auto take_bf = [&](size_t n) { __bf16* r = (__bf16*)p; p += n * sizeof(__bf16); return r; };
    __bf16* fm_in_bf  = take_bf((size_t)2 * D_INNER * D_MODEL);
    __bf16* bm_in_bf  = take_bf((size_t)2 * D_INNER * D_MODEL);
    __bf16* fm_xp_bf  = take_bf((size_t)64 * D_INNER);
    __bf16* bm_xp_bf  = take_bf((size_t)64 * D_INNER);
    __bf16* fm_dt_bf  = take_bf((size_t)D_INNER * DT_RANK);
    __bf16* bm_dt_bf  = take_bf((size_t)D_INNER * DT_RANK);
    __bf16* fm_out_bf = take_bf((size_t)D_MODEL * D_INNER);
    __bf16* bm_out_bf = take_bf((size_t)D_MODEL * D_INNER);
    __bf16* ffW1_bf   = take_bf((size_t)2048 * D_MODEL);
    __bf16* ffW2_bf   = take_bf((size_t)D_MODEL * 2048);

    __bf16* x_bf   = take_bf((size_t)ROWS * D_MODEL);
    __bf16* xrev   = take_bf((size_t)ROWS * D_MODEL);
    __bf16* xz     = take_bf((size_t)ROWS * 2 * D_INNER);
    __bf16* xm     = take_bf((size_t)ROWS * D_INNER);
    __bf16* dbc    = take_bf((size_t)ROWS * 64);
    __bf16* delta  = take_bf((size_t)ROWS * D_INNER);
    __bf16* ybuf   = take_bf((size_t)ROWS * D_INNER);
    __bf16* dirF   = take_bf((size_t)ROWS * D_MODEL);
    __bf16* dirB   = take_bf((size_t)ROWS * D_MODEL);
    // Aliased scratch (lifetimes do not overlap):
    __bf16* dirT  = xm;     // bwd out-proj result before time flip
    __bf16* ansum = delta;  // an1 + an2
    __bf16* hff   = xz;     // FFN hidden
    __bf16* ffout = ybuf;   // FFN output

    // ---------------- cast weights + input to bf16 ----------------
    auto cast = [&](const float* s, __bf16* d, int n) {
        int n4 = n / 4;
        k_cast4<<<(n4 + 255) / 256, 256, 0, stream>>>(s, d, n4);
    };
    cast(fm_in_W,  fm_in_bf,  2 * D_INNER * D_MODEL);
    cast(bm_in_W,  bm_in_bf,  2 * D_INNER * D_MODEL);
    cast(fm_xp_W,  fm_xp_bf,  64 * D_INNER);
    cast(bm_xp_W,  bm_xp_bf,  64 * D_INNER);
    cast(fm_dt_W,  fm_dt_bf,  D_INNER * DT_RANK);
    cast(bm_dt_W,  bm_dt_bf,  D_INNER * DT_RANK);
    cast(fm_out_W, fm_out_bf, D_MODEL * D_INNER);
    cast(bm_out_W, bm_out_bf, D_MODEL * D_INNER);
    cast(ff_W1,    ffW1_bf,   2048 * D_MODEL);
    cast(ff_W2,    ffW2_bf,   D_MODEL * 2048);
    cast(x,        x_bf,      ROWS * D_MODEL);

    auto gemm = [&](const __bf16* A, int lda, const __bf16* W, const float* bias,
                    __bf16* D, int N, int K, int act) {
        dim3 grid(N / BN, ROWS / BM);
        k_gemm<<<grid, 256, 0, stream>>>(A, lda, W, bias, D, N, K, act);
    };

    // One Mamba direction (input u -> out_dir, in input's time order).
    auto run_dir = [&](const __bf16* u, const __bf16* inW, const float* cw, const float* cb,
                       const __bf16* xpW, const __bf16* dtW, const float* dtb,
                       const float* Alog, const float* Dp, const __bf16* outW,
                       __bf16* out_dir) {
        gemm(u, D_MODEL, inW, nullptr, xz, 2 * D_INNER, D_MODEL, 0);           // xz
        k_conv_silu<<<(ROWS * D_INNER) / 256, 256, 0, stream>>>(xz, cw, cb, xm);
        gemm(xm, D_INNER, xpW, nullptr, dbc, 64, D_INNER, 0);                  // dt|B|C
        gemm(dbc, 64, dtW, dtb, delta, D_INNER, DT_RANK, 2);                   // softplus
        k_scan<<<dim3(D_INNER / 256, BATCH), 256, 0, stream>>>(
            delta, dbc, xm, xz, Alog, Dp, ybuf);
        gemm(ybuf, D_INNER, outW, nullptr, out_dir, D_MODEL, D_INNER, 0);      // out-proj
    };

    // Forward branch.
    run_dir(x_bf, fm_in_bf, fm_cw, fm_cb, fm_xp_bf, fm_dt_bf, fm_dt_b,
            fm_Alog, fm_D, fm_out_bf, dirF);

    // Backward branch: flip time, run, flip back.
    k_flip512<<<(ROWS * D_MODEL / 8) / 256, 256, 0, stream>>>(x_bf, xrev);
    run_dir(xrev, bm_in_bf, bm_cw, bm_cb, bm_xp_bf, bm_dt_bf, bm_dt_b,
            bm_Alog, bm_D, bm_out_bf, dirT);
    k_flip512<<<(ROWS * D_MODEL / 8) / 256, 256, 0, stream>>>(dirT, dirB);

    // an1 + an2 (bf16 accumulate into ansum).
    k_add_ln<<<ROWS, 256, 0, stream>>>(x_bf, dirF, ln1_g, ln1_b, ansum, nullptr, 0);
    k_add_ln<<<ROWS, 256, 0, stream>>>(x_bf, dirB, ln2_g, ln2_b, ansum, nullptr, 1);

    // FFN.
    gemm(ansum, D_MODEL, ffW1_bf, ff_b1, hff, 2048, D_MODEL, 1);   // ReLU
    gemm(hff, 2048, ffW2_bf, ff_b2, ffout, D_MODEL, 2048, 0);

    // Final LN(ff + ff) -> fp32 d_out.
    k_add_ln<<<ROWS, 256, 0, stream>>>(ffout, ffout, ln3_g, ln3_b, nullptr,
                                       (float*)d_out, 0);

    (void)in_sizes; (void)n_in; (void)out_size; (void)ws_size;
}
